// CausalSelfAttention_24644522345261
// MI455X (gfx1250) — matrix operand order repaired and checked by reading, not compiled, not run on hardware
//
#include <hip/hip_runtime.h>

typedef __bf16 bf16;
typedef __bf16 v16bf __attribute__((ext_vector_type(16)));
typedef __bf16 v8bf  __attribute__((ext_vector_type(8)));
typedef __bf16 v4bf  __attribute__((ext_vector_type(4)));
typedef float  v8f   __attribute__((ext_vector_type(8)));
typedef float  v4f   __attribute__((ext_vector_type(4)));
typedef unsigned int u32x4 __attribute__((ext_vector_type(4)));
typedef int          i32x8 __attribute__((ext_vector_type(8)));
typedef int          i32x4 __attribute__((ext_vector_type(4)));

#define LDSS 40   // LDS row stride in bf16 (64B data + 16B pad -> TDM pad 16DW+4DW)

static __device__ __forceinline__ v16bf cat16(v8bf lo, v8bf hi) {
  return __builtin_shufflevector(lo, hi, 0,1,2,3,4,5,6,7,8,9,10,11,12,13,14,15);
}
// A-fragment: lane holds row, K runs [8g..8g+7] and [16+8g..16+8g+7]
static __device__ __forceinline__ v16bf frag_a(const bf16* base, int g) {
  v8bf lo = *(const v8bf*)(base + 8 * g);
  v8bf hi = *(const v8bf*)(base + 16 + 8 * g);
  return cat16(lo, hi);
}
// B-fragment (operand stored transposed: row = column n, contiguous K): K runs [8g..8g+7] and [16+8g..16+8g+7]
static __device__ __forceinline__ v16bf frag_b(const bf16* base, int g) {
  v8bf lo = *(const v8bf*)(base + 8 * g);
  v8bf hi = *(const v8bf*)(base + 16 + 8 * g);
  return cat16(lo, hi);
}
static __device__ __forceinline__ v8f wmma_bf16(v16bf a, v16bf b, v8f c) {
  return __builtin_amdgcn_wmma_f32_16x16x32_bf16(false, a, false, b, (short)0, c, false, false);
}
static __device__ __forceinline__ float xswap16(float v) {   // lane ^ 16 exchange
  return __builtin_bit_cast(float, __builtin_amdgcn_ds_swizzle(__builtin_bit_cast(int, v), 0x401f));
}

// ---------------------------------------------------------------------------
// TDM: DMA a 2D bf16 tile (tile_d0 x tile_d1, row stride stride0 elems) from
// global into LDS at lds_off, padding 4 DWORDs after every 16 DWORDs so each
// 32-elem row lands on an 80-byte (LDSS=40) pitch. D# per ISA 08 §8.
// ---------------------------------------------------------------------------
static __device__ __forceinline__ void tdm_load_2d(
    unsigned int lds_off, const bf16* gptr,
    unsigned int tile_d0, unsigned int tile_d1,
    unsigned int tensor_d0, unsigned int tensor_d1,
    unsigned int stride0)
{
  unsigned long long ga = (unsigned long long)(size_t)gptr;
  u32x4 g0;
  g0[0] = 1u;                                             // count=1 valid user D#
  g0[1] = lds_off;                                        // lds_addr (bytes)
  g0[2] = (unsigned int)ga;                               // global_addr[31:0]
  g0[3] = (unsigned int)((ga >> 32) & 0x01FFFFFFu) | 0x80000000u; // [56:32] | type=2
  i32x8 g1;
  g1[0] = (int)((1u << 16)      // data_size = 1 -> 2 bytes
              | (1u << 20)      // pad_enable
              | (3u << 22)      // pad_interval code 3 -> 16 DWORDs
              | (3u << 25));    // pad_amount  code 3 -> 4 DWORDs
  g1[1] = (int)((tensor_d0 & 0xFFFFu) << 16);                       // tensor_dim0[15:0]
  g1[2] = (int)((tensor_d0 >> 16) | ((tensor_d1 & 0xFFFFu) << 16)); // td0 hi | td1 lo
  g1[3] = (int)((tensor_d1 >> 16) | (tile_d0 << 16));               // td1 hi | tile_dim0
  g1[4] = (int)(tile_d1 & 0xFFFFu);                                 // tile_dim1 (dim2=0)
  g1[5] = (int)stride0;                                             // dim0 stride lo
  g1[6] = 0;
  g1[7] = 0;
  i32x4 z4 = {0, 0, 0, 0};
#if defined(__clang_major__) && (__clang_major__ >= 23)
  i32x8 z8 = {0, 0, 0, 0, 0, 0, 0, 0};
  __builtin_amdgcn_tensor_load_to_lds(g0, g1, z4, z4, z8, 0);
#else
  __builtin_amdgcn_tensor_load_to_lds(g0, g1, z4, z4, 0);
#endif
}

// ---------------------------------------------------------------------------
// Pre-pass A: fp32 -> bf16 copy (for x)
// ---------------------------------------------------------------------------
__global__ __launch_bounds__(256) void cvt_bf16_kernel(const float* __restrict__ in,
                                                       bf16* __restrict__ out) {
  size_t i = ((size_t)blockIdx.x * 256 + threadIdx.x) * 4;
  v4f v = *(const v4f*)(in + i);
  v4bf o = { (bf16)v.x, (bf16)v.y, (bf16)v.z, (bf16)v.w };
  *(v4bf*)(out + i) = o;
}

// ---------------------------------------------------------------------------
// Pre-pass B: out[n][k] = (bf16) in[k][n]   (in: K x N fp32)
// ---------------------------------------------------------------------------
__global__ __launch_bounds__(256) void transpose_cvt_kernel(const float* __restrict__ in,
                                                            bf16* __restrict__ out,
                                                            int K, int N) {
  __shared__ float tile[32][33];
  const int k0 = blockIdx.x * 32, n0 = blockIdx.y * 32;
  const int tx = threadIdx.x & 31, ty = threadIdx.x >> 5;   // 32x8
  #pragma unroll
  for (int i = 0; i < 32; i += 8)
    tile[ty + i][tx] = in[(size_t)(k0 + ty + i) * N + n0 + tx];
  __syncthreads();
  #pragma unroll
  for (int i = 0; i < 32; i += 8)
    out[(size_t)(n0 + ty + i) * K + k0 + tx] = (bf16)tile[tx][ty + i];
}

// ---------------------------------------------------------------------------
// Kernel 1: qkv = xb(bf16) @ WqkvT(bf16)^T + bqkv ; scatter to q/k [B,H,T,64], vT [B,H,64,T]
// M=8192 K=1024 N=3072. 128x128 tile, 8 waves (256 thr), 2x4 WMMA frags/wave,
// both tiles DMA'd by the TDM each K-step.
// ---------------------------------------------------------------------------
__global__ __launch_bounds__(256) void qkv_gemm_kernel(
    const bf16* __restrict__ A, const bf16* __restrict__ Bt, const float* __restrict__ bias,
    bf16* __restrict__ qb, bf16* __restrict__ kb, bf16* __restrict__ vT)
{
  const int K = 1024, T = 2048, H = 16;
  __shared__ bf16 As[128 * LDSS];
  __shared__ bf16 Bs[128 * LDSS];
  const int tid  = threadIdx.x;
  const int lane = tid & 31, w = tid >> 5;
  const int g = lane >> 4, ln = lane & 15;
  const int wm = (w >> 1) * 32, wn = (w & 1) * 64;
  const int m0 = blockIdx.x * 128, n0 = blockIdx.y * 128;
  const unsigned int asOff = (unsigned int)(size_t)(void*)As;
  const unsigned int bsOff = (unsigned int)(size_t)(void*)Bs;

  v8f acc[2][4] = {};

  for (int k0 = 0; k0 < K; k0 += 32) {
    __syncthreads();
    if (w == 0) {
      tdm_load_2d(asOff, A  + (size_t)m0 * K + k0, 32, 128, K, 8192, K);
      tdm_load_2d(bsOff, Bt + (size_t)n0 * K + k0, 32, 128, K, 3072, K);
      __builtin_amdgcn_s_wait_tensorcnt(0);
    }
    __syncthreads();
    v16bf a[2], b[4];
    #pragma unroll
    for (int fm = 0; fm < 2; ++fm) a[fm] = frag_a(As + (wm + fm * 16 + ln) * LDSS, g);
    #pragma unroll
    for (int fn = 0; fn < 4; ++fn) b[fn] = frag_b(Bs + (wn + fn * 16 + ln) * LDSS, g);
    #pragma unroll
    for (int fm = 0; fm < 2; ++fm)
      #pragma unroll
      for (int fn = 0; fn < 4; ++fn)
        acc[fm][fn] = wmma_bf16(a[fm], b[fn], acc[fm][fn]);
  }

  #pragma unroll
  for (int fm = 0; fm < 2; ++fm)
    #pragma unroll
    for (int fn = 0; fn < 4; ++fn) {
      int col3 = n0 + wn + fn * 16 + ln;       // 0..3071
      float bv = bias[col3];
      int which = col3 >> 10;                  // 0=q 1=k 2=v
      int cmod  = col3 & 1023;
      int h = cmod >> 6, dd = cmod & 63;
      int rowb = m0 + wm + fm * 16 + 8 * g;    // rows rowb..rowb+7 (same batch)
      int b_ = rowb >> 11;
      int t = rowb & 2047;
      size_t bh = (size_t)b_ * H + h;
      if (which == 2) {
        v8bf o;
        #pragma unroll
        for (int r = 0; r < 8; ++r) o[r] = (bf16)(acc[fm][fn][r] + bv);
        *(v8bf*)(vT + (bh * 64 + dd) * T + t) = o;   // contiguous in t
      } else {
        bf16* dst = (which == 0) ? qb : kb;
        #pragma unroll
        for (int r = 0; r < 8; ++r)
          dst[(bh * T + t + r) * 64 + dd] = (bf16)(acc[fm][fn][r] + bv);
      }
    }
}

// ---------------------------------------------------------------------------
// Kernel 2: transposed flash attention (wave = 16 queries; S^T = K*Q^T so the
// softmax reduction is lane-local + one SWAPX16; y^T = V^T * P^T).
// ---------------------------------------------------------------------------
__global__ __launch_bounds__(256) void attn_kernel(
    const bf16* __restrict__ q, const bf16* __restrict__ k,
    const bf16* __restrict__ vT, bf16* __restrict__ y)
{
  const int T = 2048, D = 64, C = 1024;
  __shared__ bf16 Pl[8 * 16 * LDSS];
  const int lane = threadIdx.x & 31, w = threadIdx.x >> 5;
  const int g = lane >> 4, ln = lane & 15;
  const int bh = blockIdx.x;                   // b*16 + h
  const int b = bh >> 4, h = bh & 15;
  const int q0 = w * 256 + blockIdx.y * 16;    // balanced causal work per block
  const bf16* qb = q  + (size_t)bh * T * D;
  const bf16* kb = k  + (size_t)bh * T * D;
  const bf16* vb = vT + (size_t)bh * D * T;
  bf16* pw = Pl + w * 16 * LDSS;

  const bf16* qrow = qb + (size_t)(q0 + ln) * D;
  v16bf Bq0 = frag_b(qrow, g);
  v16bf Bq1 = frag_b(qrow + 32, g);

  float mi = -3.0e38f, li = 0.0f;
  v8f a0 = {}, a1 = {}, a2 = {}, a3 = {};
  const float scale = 0.125f;                  // 1/sqrt(64)
  const int kend = q0 + 16;                    // causal bound (wave-uniform)

  for (int k0 = 0; k0 < kend; k0 += 32) {
    v8f st[2];
    #pragma unroll
    for (int s = 0; s < 2; ++s) {
      const bf16* krow = kb + (size_t)(k0 + s * 16 + ln) * D;
      v16bf Ak0 = frag_a(krow, g);
      v16bf Ak1 = frag_a(krow + 32, g);
      v8f z = {};
      z = wmma_bf16(Ak0, Bq0, z);
      z = wmma_bf16(Ak1, Bq1, z);
      st[s] = z;
    }
    float sv[16];
    float tmax = -3.0e38f;
    #pragma unroll
    for (int s = 0; s < 2; ++s)
      #pragma unroll
      for (int r = 0; r < 8; ++r) {
        int key = k0 + s * 16 + 8 * g + r;
        float vv = st[s][r] * scale;
        vv = (key > q0 + ln) ? -3.0e38f : vv;
        sv[s * 8 + r] = vv;
        tmax = fmaxf(tmax, vv);
      }
    tmax = fmaxf(tmax, xswap16(tmax));
    float mnew  = fmaxf(mi, tmax);
    float alpha = __expf(mi - mnew);
    float ps = 0.0f;
    #pragma unroll
    for (int i = 0; i < 16; ++i) { float p = __expf(sv[i] - mnew); sv[i] = p; ps += p; }
    ps += xswap16(ps);
    li = li * alpha + ps;
    mi = mnew;
    a0 *= alpha; a1 *= alpha; a2 *= alpha; a3 *= alpha;

    bf16* prow = pw + ln * LDSS;
    v8bf p0, p1;
    #pragma unroll
    for (int r = 0; r < 8; ++r) { p0[r] = (bf16)sv[r]; p1[r] = (bf16)sv[8 + r]; }
    *(v8bf*)(prow + 8 * g)      = p0;
    *(v8bf*)(prow + 16 + 8 * g) = p1;
    v16bf Bp = frag_b(prow, g);

    #pragma unroll
    for (int f = 0; f < 4; ++f) {
      const bf16* vrow = vb + (size_t)(f * 16 + ln) * T + k0;
      v16bf Av = frag_a(vrow, g);
      v8f& acc = (f == 0) ? a0 : (f == 1) ? a1 : (f == 2) ? a2 : a3;
      acc = wmma_bf16(Av, Bp, acc);
    }
  }

  float inv = 1.0f / li;
  bf16* yrow = y + ((size_t)b * T + q0 + ln) * C + h * 64;
  v8f accs[4] = {a0, a1, a2, a3};
  #pragma unroll
  for (int f = 0; f < 4; ++f) {
    v8bf o;
    #pragma unroll
    for (int r = 0; r < 8; ++r) o[r] = (bf16)(accs[f][r] * inv);
    *(v8bf*)(yrow + f * 16 + 8 * g) = o;
  }
}

// ---------------------------------------------------------------------------
// Kernel 3: out = y(bf16) @ WprojT(bf16)^T + bproj, fp32 out. Same TDM engine.
// ---------------------------------------------------------------------------
__global__ __launch_bounds__(256) void proj_gemm_kernel(
    const bf16* __restrict__ A, const bf16* __restrict__ Bt, const float* __restrict__ bias,
    float* __restrict__ out)
{
  const int K = 1024, N = 1024;
  __shared__ bf16 As[128 * LDSS];
  __shared__ bf16 Bs[128 * LDSS];
  const int tid  = threadIdx.x;
  const int lane = tid & 31, w = tid >> 5;
  const int g = lane >> 4, ln = lane & 15;
  const int wm = (w >> 1) * 32, wn = (w & 1) * 64;
  const int m0 = blockIdx.x * 128, n0 = blockIdx.y * 128;
  const unsigned int asOff = (unsigned int)(size_t)(void*)As;
  const unsigned int bsOff = (unsigned int)(size_t)(void*)Bs;

  v8f acc[2][4] = {};

  for (int k0 = 0; k0 < K; k0 += 32) {
    __syncthreads();
    if (w == 0) {
      tdm_load_2d(asOff, A  + (size_t)m0 * K + k0, 32, 128, K, 8192, K);
      tdm_load_2d(bsOff, Bt + (size_t)n0 * K + k0, 32, 128, K, 1024, K);
      __builtin_amdgcn_s_wait_tensorcnt(0);
    }
    __syncthreads();
    v16bf a[2], b[4];
    #pragma unroll
    for (int fm = 0; fm < 2; ++fm) a[fm] = frag_a(As + (wm + fm * 16 + ln) * LDSS, g);
    #pragma unroll
    for (int fn = 0; fn < 4; ++fn) b[fn] = frag_b(Bs + (wn + fn * 16 + ln) * LDSS, g);
    #pragma unroll
    for (int fm = 0; fm < 2; ++fm)
      #pragma unroll
      for (int fn = 0; fn < 4; ++fn)
        acc[fm][fn] = wmma_bf16(a[fm], b[fn], acc[fm][fn]);
  }

  #pragma unroll
  for (int fm = 0; fm < 2; ++fm)
    #pragma unroll
    for (int fn = 0; fn < 4; ++fn) {
      int col = n0 + wn + fn * 16 + ln;
      float bv = bias[col];
      int row = m0 + wm + fm * 16 + 8 * g;
      #pragma unroll
      for (int r = 0; r < 8; ++r)
        out[(size_t)(row + r) * N + col] = acc[fm][fn][r] + bv;
    }
}

// ---------------------------------------------------------------------------
extern "C" void kernel_launch(void* const* d_in, const int* in_sizes, int n_in,
                              void* d_out, int out_size, void* d_ws, size_t ws_size,
                              hipStream_t stream) {
  (void)in_sizes; (void)n_in; (void)out_size; (void)ws_size;
  const float* x     = (const float*)d_in[0];   // [4,2048,1024]
  const float* Wqkv  = (const float*)d_in[1];   // [1024,3072]
  const float* bqkv  = (const float*)d_in[2];   // [3072]
  const float* Wproj = (const float*)d_in[3];   // [1024,1024]
  const float* bproj = (const float*)d_in[4];   // [1024]
  float* out = (float*)d_out;                   // [4,2048,1024]

  char* ws = (char*)d_ws;
  const size_t SEG = (size_t)16 * 1024 * 1024;
  bf16* qb  = (bf16*)(ws + 0 * SEG);            // [B,H,T,64]
  bf16* kb  = (bf16*)(ws + 1 * SEG);            // [B,H,T,64]
  bf16* vT  = (bf16*)(ws + 2 * SEG);            // [B,H,64,T]
  bf16* yb  = (bf16*)(ws + 3 * SEG);            // [B,T,C]
  bf16* xb  = (bf16*)(ws + 4 * SEG);            // x as bf16 [8192,1024]
  bf16* wqt = (bf16*)(ws + 5 * SEG);            // Wqkv^T bf16 [3072,1024] (6MB)
  bf16* wpt = (bf16*)(ws + 5 * SEG + (size_t)8 * 1024 * 1024); // Wproj^T bf16 [1024,1024]

  cvt_bf16_kernel     <<<8192, 256, 0, stream>>>(x, xb);             // 8M elems / 4
  transpose_cvt_kernel<<<dim3(32, 96), 256, 0, stream>>>(Wqkv, wqt, 1024, 3072);
  transpose_cvt_kernel<<<dim3(32, 32), 256, 0, stream>>>(Wproj, wpt, 1024, 1024);
  qkv_gemm_kernel     <<<dim3(64, 24), 256, 0, stream>>>(xb, wqt, bqkv, qb, kb, vT);
  attn_kernel         <<<dim3(64, 16), 256, 0, stream>>>(qb, kb, vT, yb);
  proj_gemm_kernel    <<<dim3(64, 8),  256, 0, stream>>>(yb, wpt, bproj, out);
}
